// NeuralGraph_61503931678807
// MI455X (gfx1250) — matrix operand order repaired and checked by reading, not compiled, not run on hardware
//
#include <hip/hip_runtime.h>
#include <hip/hip_bf16.h>
#include <math.h>

typedef __attribute__((ext_vector_type(16))) _Float16 v16h;
typedef __attribute__((ext_vector_type(8)))  float    v8f;

constexpr int B  = 2;
constexpr int NN = 50000;
constexpr int NE = 800000;
constexpr float ATT_SCALE = 0.25f;   // 1/sqrt(16)

// ---- workspace layout (bytes) ----
constexpr size_t OFF_W   = 0;                                   // 40 frags * 512 f16 = 40KB
constexpr size_t OFF_ATT = 64 * 1024;                           // f32 [B*NN*64]
constexpr size_t OFF_MAB = OFF_ATT + (size_t)B * NN * 64 * 4;   // f16 [B*NE*32]
constexpr size_t OFF_LOG = OFF_MAB + (size_t)B * NE * 32 * 2;   // f32 [2*B*NE]
constexpr size_t OFF_MAX = OFF_LOG + (size_t)2 * B * NE * 4;    // u32 [2*B*NN]
constexpr size_t OFF_DEN = OFF_MAX + (size_t)2 * B * NN * 4;    // f32 [2*B*NN]
constexpr size_t OFF_NUM = OFF_DEN + (size_t)2 * B * NN * 4;    // f32 [2*B*NN*16]

// 9 weight matrices: msg(48x64,64x32,32x48) upd(48x64,64x32,32x16) att(16x64,64x32,32x64)
__constant__ int dW_K[9]    = {48,64,32, 48,64,32, 16,64,32};
__constant__ int dW_N[9]    = {64,32,48, 64,32,16, 64,32,64};
__constant__ int dW_NT[9]   = {4,2,3, 4,2,1, 4,2,4};
__constant__ int dW_CNT[9]  = {8,4,3, 8,4,1, 4,4,4};
__constant__ int dW_BASE[9] = {0,8,12, 15,23,27, 28,32,36};
// fragment bases for use sites (compile-time)
constexpr int FB_MSG1 = 0, FB_MSG2 = 8,  FB_MSG3 = 12;
constexpr int FB_UPD1 = 15, FB_UPD2 = 23, FB_UPD3 = 27;
constexpr int FB_ATT1 = 28, FB_ATT2 = 32, FB_ATT3 = 36;

__device__ __forceinline__ void ldsfence() {
  asm volatile("s_wait_dscnt 0" ::: "memory");
}

__device__ __forceinline__ unsigned fenc(float x) {
  unsigned u = __float_as_uint(x);
  return (u & 0x80000000u) ? ~u : (u | 0x80000000u);
}
__device__ __forceinline__ float fdec(unsigned e) {
  unsigned u = (e & 0x80000000u) ? (e & 0x7FFFFFFFu) : ~e;
  return __uint_as_float(u);
}

// gfx1250 transcendental tanh; v_nop covers the 1-op TRANS result hazard
__device__ __forceinline__ float soft_clamp(float x) {
  float r;
  asm("v_tanh_f32 %0, %1\n\tv_nop" : "=v"(r) : "v"(x * 1e-6f));
  return r * 1e6f;
}

// A fragment (16x32 f16) from row-major LDS tile, stride in halves.
// Layout (ISA 7.12.2): lanes 0-15 row M=lane, halves {K=0..7,16..23};
// lanes 16-31 same rows, halves {K=8..15,24..31}.
__device__ __forceinline__ v16h lds_loadA(const _Float16* x, int lane, int kchunk, int stride) {
  int row = lane & 15;
  int hi  = (lane >> 4) << 3;   // 0 or 8
  const _Float16* p = x + row * stride + kchunk * 32 + hi;
  v16h a;
#pragma unroll
  for (int j = 0; j < 8; ++j) { a[j] = p[j]; a[j + 8] = p[16 + j]; }
  return a;
}

// One MLP layer: D(16 x NT*16) = X(16 x KC*32) * W, accumulators in v8f frags.
template <int KC, int NT>
__device__ __forceinline__ void wmma_layer(const _Float16* xin, int stride,
                                           const _Float16* wf, int fragbase,
                                           int lane, v8f* acc) {
  v16h a[KC];
#pragma unroll
  for (int k = 0; k < KC; ++k) a[k] = lds_loadA(xin, lane, k, stride);
#pragma unroll
  for (int n = 0; n < NT; ++n) {
    v8f c = {};
#pragma unroll
    for (int k = 0; k < KC; ++k) {
      v16h bb = *(const v16h*)(wf + ((size_t)(fragbase + k * NT + n) * 32 + lane) * 16);
      c = __builtin_amdgcn_wmma_f32_16x16x32_f16(false, a[k], false, bb,
                                                 (short)0, c, false, false);
    }
    acc[n] = c;
  }
}

// bias + SiLU + store back to LDS as f16 (C/D layout: m = r + 8*(lane>=16), n = lane&15)
template <int NT>
__device__ __forceinline__ void silu_store(const v8f* acc, const float* bias,
                                           _Float16* xout, int stride, int lane) {
  int n  = lane & 15;
  int rb = (lane >> 4) << 3;
#pragma unroll
  for (int t = 0; t < NT; ++t) {
    float bn = bias[t * 16 + n];
#pragma unroll
    for (int r = 0; r < 8; ++r) {
      float h = acc[t][r] + bn;
      float s = h * __builtin_amdgcn_rcpf(1.0f + __expf(-h));
      xout[(rb + r) * stride + t * 16 + n] = (_Float16)s;
    }
  }
}

// ---------------- weight repack: f32 (K,N) row-major -> f16 B-fragments ----------------
struct WPtrs { const float* p[9]; };

__global__ void kw(WPtrs wp, _Float16* wf) {
  int blk = blockIdx.x;              // 0..39 : global fragment id
  int m = 0;
  while (m < 8 && blk >= dW_BASE[m] + dW_CNT[m]) ++m;
  int f  = blk - dW_BASE[m];
  int NT = dW_NT[m];
  int kc = f / NT, nt = f % NT;
  int t = threadIdx.x;               // 0..511
  int lane = t >> 4, j = t & 15;
  int k = kc * 32 + ((j & 8) << 1) + ((lane >= 16) ? 8 : 0) + (j & 7);
  int n = nt * 16 + (lane & 15);
  float v = 0.f;
  if (k < dW_K[m]) v = wp.p[m][(size_t)k * dW_N[m] + n];
  wf[(size_t)blk * 512 + (size_t)lane * 16 + j] = (_Float16)v;
}

// ---------------- init accumulators ----------------
__global__ void kinit(unsigned* maxenc, float* den, float* num) {
  int i = blockIdx.x * 256 + threadIdx.x;
  const int NMX = 2 * B * NN;            // 200000
  const int NNUM = 2 * B * NN * 16;      // 3.2M
  if (i < NMX) { maxenc[i] = 0x00800000u; den[i] = 0.f; }
  if (i < NNUM) num[i] = 0.f;
}

// ---------------- attention MLP: 16 -> 64 -> 32 -> 64 ----------------
__global__ __launch_bounds__(64) void katt(const float* nodes, const float* b0,
                                           const float* b1, const float* b2,
                                           const _Float16* wf, float* attw) {
  __shared__ _Float16 xb[2][16 * 64];
  int w = threadIdx.x >> 5, lane = threadIdx.x & 31;
  _Float16* X = xb[w];
  int tile = blockIdx.x * 2 + w;          // 0..6249
  int b = tile / (NN / 16);
  int node0 = (tile % (NN / 16)) * 16;

  // zero pad cols 16..31 (layer1 K-chunk 0 covers cols 0..31 only)
  for (int i = lane; i < 256; i += 32) {
    int r = i >> 4, c = i & 15;
    X[r * 64 + 16 + c] = (_Float16)0.f;
  }
  // contiguous 16x16 node tile
  {
    const float* src = nodes + ((size_t)b * NN + node0) * 16;
    for (int i = lane; i < 256; i += 32) {
      int r = i >> 4, c = i & 15;
      X[r * 64 + c] = (_Float16)src[i];
    }
  }
  ldsfence();

  v8f acc[4];
  wmma_layer<1, 4>(X, 64, wf, FB_ATT1, lane, acc);
  silu_store<4>(acc, b0, X, 64, lane);
  ldsfence();
  wmma_layer<2, 2>(X, 64, wf, FB_ATT2, lane, acc);
  silu_store<2>(acc, b1, X, 64, lane);
  ldsfence();
  wmma_layer<1, 4>(X, 64, wf, FB_ATT3, lane, acc);

  int n = lane & 15, rb = (lane >> 4) << 3;
#pragma unroll
  for (int t = 0; t < 4; ++t) {
    float bn = b2[t * 16 + n];
#pragma unroll
    for (int r = 0; r < 8; ++r)
      attw[((size_t)b * NN + node0 + rb + r) * 64 + t * 16 + n] = acc[t][r] + bn;
  }
}

// ---------------- message MLP + logits + atomic max + edge output ----------------
__global__ __launch_bounds__(64) void kmsg(const float* nodes, const float* edges,
                                           const int* ca, const int* cb,
                                           const float* b0, const float* b1, const float* b2,
                                           const _Float16* wf, const float* attw,
                                           float* logit, unsigned* maxenc,
                                           _Float16* mab, float* out_edges) {
  __shared__ _Float16 xb[2][16 * 64];
  __shared__ int cA[2][16], cB[2][16];
  int w = threadIdx.x >> 5, lane = threadIdx.x & 31;
  _Float16* X = xb[w];
  int tile = blockIdx.x * 2 + w;          // 0..99999
  int b = tile / (NE / 16);
  int e0 = (tile % (NE / 16)) * 16;
  int r16 = lane & 15;

  // each lane owns one conn index: lanes 0-15 -> conn_a, 16-31 -> conn_b
  int myconn = (lane < 16) ? ca[e0 + r16] : cb[e0 + r16];
  if (lane < 16) cA[w][r16] = myconn;
  else           cB[w][r16] = myconn;

  // prefetch next edge tiles (speculative, CDNA5 global_prefetch_b8)
  __builtin_prefetch(edges + ((size_t)b * NE + e0 + 256) * 16, 0, 3);

  // zero pad cols 48..63
  for (int i = lane; i < 256; i += 32) {
    int r = i >> 4, c = i & 15;
    X[r * 64 + 48 + c] = (_Float16)0.f;
  }
  // na (cols 0..15) / nb (cols 16..31): each lane fills one row from its own conn
  {
    const float* src = nodes + ((size_t)b * NN + myconn) * 16;
    int cbase = (lane < 16) ? 0 : 16;
#pragma unroll
    for (int c = 0; c < 16; ++c) X[r16 * 64 + cbase + c] = (_Float16)src[c];
  }
  // edge features (cols 32..47): fully contiguous 1KB tile
  {
    const float* esrc = edges + ((size_t)b * NE + e0) * 16;
    for (int i = lane; i < 256; i += 32) {
      int r = i >> 4, c = i & 15;
      X[r * 64 + 32 + c] = (_Float16)esrc[i];
    }
  }
  ldsfence();

  // logits: lanes 0-15 dir0 (logit_b = f_q[nb].f_k[na], seg conn_b);
  //         lanes 16-31 dir1 (logit_a = b_q[na].b_k[nb], seg conn_a)
  {
    int na = cA[w][r16], nb_ = cB[w][r16];
    const float *Q, *A;
    int dir, segnode;
    if (lane < 16) { Q = attw + ((size_t)b * NN + nb_) * 64 + 16;
                     A = attw + ((size_t)b * NN + na) * 64 + 0;  dir = 0; segnode = nb_; }
    else           { Q = attw + ((size_t)b * NN + na) * 64 + 48;
                     A = attw + ((size_t)b * NN + nb_) * 64 + 32; dir = 1; segnode = na; }
    float s = 0.f;
#pragma unroll
    for (int k = 0; k < 16; ++k) s += Q[k] * A[k];
    s *= ATT_SCALE;
    logit[((size_t)dir * B + b) * NE + e0 + r16] = s;
    atomicMax(&maxenc[((size_t)dir * B + b) * NN + segnode], fenc(s));
  }

  v8f acc[4];
  wmma_layer<2, 4>(X, 64, wf, FB_MSG1, lane, acc);
  silu_store<4>(acc, b0, X, 64, lane);
  ldsfence();
  wmma_layer<2, 2>(X, 64, wf, FB_MSG2, lane, acc);
  silu_store<2>(acc, b1, X, 64, lane);
  ldsfence();
  wmma_layer<1, 3>(X, 64, wf, FB_MSG3, lane, acc);

  int n = lane & 15, rb = (lane >> 4) << 3;
#pragma unroll
  for (int t = 0; t < 2; ++t) {              // m_a (t=0), m_b (t=1) -> ws f16
    float bn = b2[t * 16 + n];
#pragma unroll
    for (int r = 0; r < 8; ++r)
      mab[((size_t)b * NE + e0 + rb + r) * 32 + t * 16 + n] = (_Float16)(acc[t][r] + bn);
  }
  {                                          // m_e -> new_edges
    float bn = b2[32 + n];
#pragma unroll
    for (int r = 0; r < 8; ++r) {
      size_t idx = ((size_t)b * NE + e0 + rb + r) * 16 + n;
      out_edges[idx] = soft_clamp(edges[idx] + acc[2][r] + bn);
    }
  }
}

// ---------------- segment softmax scatter ----------------
__global__ void kscat(const int* ca, const int* cb, const float* logit,
                      const unsigned* maxenc, const _Float16* mab,
                      float* den, float* num) {
  size_t id = (size_t)blockIdx.x * 256 + threadIdx.x;   // < 2*B*NE*16
  int c = id & 15;
  int dir = (id >> 4) & 1;
  size_t rest = id >> 5;
  int e = (int)(rest % NE);
  int b = (int)(rest / NE);
  int node = (dir == 0) ? cb[e] : ca[e];
  float l  = logit[((size_t)dir * B + b) * NE + e];
  float mx = fdec(maxenc[((size_t)dir * B + b) * NN + node]);
  float wgt = __expf(l - mx);
  float m = (float)mab[((size_t)b * NE + e) * 32 + (dir == 0 ? 16 : 0) + c];
  atomicAdd(&num[(((size_t)dir * B + b) * NN + node) * 16 + c], wgt * m);
  if (c == 0) atomicAdd(&den[((size_t)dir * B + b) * NN + node], wgt);
}

// ---------------- update MLP: 48 -> 64 -> 32 -> 16, residual + clamp ----------------
__global__ __launch_bounds__(64) void kupd(const float* nodes, const float* b0,
                                           const float* b1, const float* b2,
                                           const _Float16* wf, const float* den,
                                           const float* num, float* out_nodes) {
  __shared__ _Float16 xb[2][16 * 64];
  int w = threadIdx.x >> 5, lane = threadIdx.x & 31;
  _Float16* X = xb[w];
  int tile = blockIdx.x * 2 + w;
  int b = tile / (NN / 16);
  int node0 = (tile % (NN / 16)) * 16;

  // zero pad cols 48..63
  for (int i = lane; i < 256; i += 32) {
    int r = i >> 4, c = i & 15;
    X[r * 64 + 48 + c] = (_Float16)0.f;
  }
  // nodes (cols 0..15): contiguous tile
  {
    const float* src = nodes + ((size_t)b * NN + node0) * 16;
    for (int i = lane; i < 256; i += 32) {
      int r = i >> 4, c = i & 15;
      X[r * 64 + c] = (_Float16)src[i];
    }
  }
  // agg_a (cols 16..31, dir1=conn_a) and agg_b (cols 32..47, dir0=conn_b)
  for (int i = lane; i < 512; i += 32) {
    int r = i >> 5, c = i & 31;
    int node = node0 + r;
    int dir = (c < 16) ? 1 : 0;
    int cc  = c & 15;
    size_t base = ((size_t)dir * B + b) * NN + node;
    float v = num[base * 16 + cc] * __builtin_amdgcn_rcpf(den[base] + 1e-9f);
    X[r * 64 + 16 + c] = (_Float16)v;
  }
  ldsfence();

  v8f acc[4];
  wmma_layer<2, 4>(X, 64, wf, FB_UPD1, lane, acc);
  silu_store<4>(acc, b0, X, 64, lane);
  ldsfence();
  wmma_layer<2, 2>(X, 64, wf, FB_UPD2, lane, acc);
  silu_store<2>(acc, b1, X, 64, lane);
  ldsfence();
  wmma_layer<1, 1>(X, 64, wf, FB_UPD3, lane, acc);

  int n = lane & 15, rb = (lane >> 4) << 3;
  float bn = b2[n];
#pragma unroll
  for (int r = 0; r < 8; ++r) {
    size_t idx = ((size_t)b * NN + node0 + rb + r) * 16 + n;
    out_nodes[idx] = soft_clamp(nodes[idx] + acc[0][r] + bn);
  }
}

extern "C" void kernel_launch(void* const* d_in, const int* in_sizes, int n_in,
                              void* d_out, int out_size, void* d_ws, size_t ws_size,
                              hipStream_t stream) {
  (void)in_sizes; (void)n_in; (void)out_size; (void)ws_size;
  const float* nodes = (const float*)d_in[0];
  const float* edges = (const float*)d_in[1];
  const int*   ca    = (const int*)d_in[2];
  const int*   cb    = (const int*)d_in[3];
  const float* mW0 = (const float*)d_in[4],  *mb0 = (const float*)d_in[5];
  const float* mW1 = (const float*)d_in[6],  *mb1 = (const float*)d_in[7];
  const float* mW2 = (const float*)d_in[8],  *mb2 = (const float*)d_in[9];
  const float* uW0 = (const float*)d_in[10], *ub0 = (const float*)d_in[11];
  const float* uW1 = (const float*)d_in[12], *ub1 = (const float*)d_in[13];
  const float* uW2 = (const float*)d_in[14], *ub2 = (const float*)d_in[15];
  const float* aW0 = (const float*)d_in[16], *ab0 = (const float*)d_in[17];
  const float* aW1 = (const float*)d_in[18], *ab1 = (const float*)d_in[19];
  const float* aW2 = (const float*)d_in[20], *ab2 = (const float*)d_in[21];

  char* ws = (char*)d_ws;
  _Float16* wf     = (_Float16*)(ws + OFF_W);
  float*    attw   = (float*)(ws + OFF_ATT);
  _Float16* mab    = (_Float16*)(ws + OFF_MAB);
  float*    logit  = (float*)(ws + OFF_LOG);
  unsigned* maxenc = (unsigned*)(ws + OFF_MAX);
  float*    den    = (float*)(ws + OFF_DEN);
  float*    num    = (float*)(ws + OFF_NUM);

  float* out_nodes = (float*)d_out;
  float* out_edges = out_nodes + (size_t)B * NN * 16;

  WPtrs wp;
  wp.p[0] = mW0; wp.p[1] = mW1; wp.p[2] = mW2;
  wp.p[3] = uW0; wp.p[4] = uW1; wp.p[5] = uW2;
  wp.p[6] = aW0; wp.p[7] = aW1; wp.p[8] = aW2;

  kw<<<40, 512, 0, stream>>>(wp, wf);
  kinit<<<(2 * B * NN * 16 + 255) / 256, 256, 0, stream>>>(maxenc, den, num);
  katt<<<(B * NN / 16) / 2, 64, 0, stream>>>(nodes, ab0, ab1, ab2, wf, attw);
  kmsg<<<(B * NE / 16) / 2, 64, 0, stream>>>(nodes, edges, ca, cb, mb0, mb1, mb2,
                                             wf, attw, logit, maxenc, mab, out_edges);
  kscat<<<(int)(((size_t)2 * B * NE * 16) / 256), 256, 0, stream>>>(ca, cb, logit, maxenc,
                                                                   mab, den, num);
  kupd<<<(B * NN / 16) / 2, 64, 0, stream>>>(nodes, ub0, ub1, ub2, wf, den, num, out_nodes);
}
